// CleanMelLayer_73985106640950
// MI455X (gfx1250) — matrix operand order repaired and checked by reading, not compiled, not run on hardware
//
#include <hip/hip_runtime.h>

// ---------------- problem constants (from reference) ----------------
constexpr int CB = 2, CFQ = 257, CT = 100, CH = 96;
constexpr int CS = 8, CDIN = 192, CN = 16, CR = 6, CKF = 5, CKM = 4;
constexpr int CG = 8, CHG = CH / CG;            // 12
constexpr int CBM = CB * CFQ;                   // 514 sequences
constexpr int CROWS = CBM * CT;                 // 51400 GEMM rows
constexpr int CDP = 2 * CDIN;                   // 384
constexpr int CXPD = CR + 2 * CN;               // 38
constexpr long NTOT = (long)CB * CFQ * CT * CH; // 4,934,400

typedef _Float16 v16h __attribute__((ext_vector_type(16)));
typedef _Float16 v8h  __attribute__((ext_vector_type(8)));
typedef float    v8f  __attribute__((ext_vector_type(8)));

__device__ __forceinline__ float silu_f(float v) { return v / (1.f + __expf(-v)); }

// ---------------- LayerNorm over H=96 : one wave per row ----------------
__global__ void ln_kernel(const float* __restrict__ x, const float* __restrict__ g,
                          const float* __restrict__ b, float* __restrict__ y, int rows) {
  int r = blockIdx.x;
  if (r >= rows) return;
  int lane = threadIdx.x;
  const float* xr = x + (size_t)r * CH;
  float v0 = xr[lane], v1 = xr[lane + 32], v2 = xr[lane + 64];
  float s = v0 + v1 + v2, ss = v0 * v0 + v1 * v1 + v2 * v2;
  for (int o = 16; o > 0; o >>= 1) { s += __shfl_xor(s, o, 32); ss += __shfl_xor(ss, o, 32); }
  float m = s * (1.f / CH);
  float var = ss * (1.f / CH) - m * m;
  float rstd = rsqrtf(var + 1e-5f);
  float* yr = y + (size_t)r * CH;
  yr[lane]      = (v0 - m) * rstd * g[lane]      + b[lane];
  yr[lane + 32] = (v1 - m) * rstd * g[lane + 32] + b[lane + 32];
  yr[lane + 64] = (v2 - m) * rstd * g[lane + 64] + b[lane + 64];
}

// -------- grouped conv over frequency (KF=5, groups=8) + PReLU + residual --------
__global__ void fconv_kernel(const float* __restrict__ x, const float* __restrict__ yln,
                             const float* __restrict__ cw, const float* __restrict__ cb,
                             const float* __restrict__ alpha, float* __restrict__ out) {
  int row = blockIdx.x;                // (b, f, t) linear
  int h = threadIdx.x;                 // 96
  int t = row % CT;
  int f = (row / CT) % CFQ;
  int b = row / (CT * CFQ);
  int grp = h / CHG;
  float acc = cb[h];
  for (int k = 0; k < CKF; ++k) {
    int ff = f + k - (CKF / 2);
    if (ff < 0 || ff >= CFQ) continue;
    const float* src = yln + (((size_t)(b * CFQ + ff) * CT + t) * CH + grp * CHG);
    const float* wk = cw + (size_t)h * CHG * CKF + k;
#pragma unroll
    for (int i = 0; i < CHG; ++i) acc += wk[i * CKF] * src[i];
  }
  float v = acc > 0.f ? acc : alpha[h] * acc;
  size_t idx = (size_t)row * CH + h;
  out[idx] = x[idx] + v;
}

// ---------------- _full block: squeeze / per-group FQxFQ / unsqueeze ----------------
__global__ void squeeze_kernel(const float* __restrict__ yln, const float* __restrict__ sqw,
                               const float* __restrict__ sqb, float* __restrict__ z) {
  int idx = blockIdx.x * blockDim.x + threadIdx.x;   // B*T*S*FQ
  if (idx >= CB * CT * CS * CFQ) return;
  int f = idx % CFQ; int s = (idx / CFQ) % CS; int bt = idx / (CFQ * CS);
  int t = bt % CT, b = bt / CT;
  const float* src = yln + ((size_t)(b * CFQ + f) * CT + t) * CH;
  float acc = sqb[s];
  for (int hh = 0; hh < CH; ++hh) acc += src[hh] * sqw[s * CH + hh];
  z[idx] = silu_f(acc);
}

__global__ void fullmm_kernel(const float* __restrict__ z, const float* __restrict__ fw,
                              const float* __restrict__ fb, float* __restrict__ w) {
  int idx = blockIdx.x * blockDim.x + threadIdx.x;   // B*T*S*FQ (k fastest)
  if (idx >= CB * CT * CS * CFQ) return;
  int k = idx % CFQ; int s = (idx / CFQ) % CS; int bt = idx / (CFQ * CS);
  const float* zr = z + ((size_t)bt * CS + s) * CFQ;
  const float* wr = fw + ((size_t)s * CFQ + k) * CFQ;
  float acc = fb[s * CFQ + k];
  for (int f = 0; f < CFQ; ++f) acc += zr[f] * wr[f];
  w[idx] = acc;
}

__global__ void unsqueeze_kernel(const float* __restrict__ xin, const float* __restrict__ wbuf,
                                 const float* __restrict__ uw, const float* __restrict__ ub,
                                 float* __restrict__ out) {
  long idx = (long)blockIdx.x * blockDim.x + threadIdx.x;   // NTOT (b,f,t,h)
  if (idx >= NTOT) return;
  int h = idx % CH; long r = idx / CH;
  int t = r % CT; long r2 = r / CT;
  int f = r2 % CFQ; int b = r2 / CFQ;
  float acc = ub[h];
#pragma unroll
  for (int s = 0; s < CS; ++s)
    acc += wbuf[((size_t)(b * CT + t) * CS + s) * CFQ + f] * uw[h * CS + s];
  out[idx] = xin[idx] + silu_f(acc);
}

// ---------------- f32 -> f16 helpers ----------------
__global__ void cvt16_kernel(const float* __restrict__ src, _Float16* __restrict__ dst, long n) {
  long i = (long)blockIdx.x * blockDim.x + threadIdx.x;
  if (i < n) dst[i] = (_Float16)src[i];
}

__global__ void prep_inA_kernel(const float* __restrict__ xn, _Float16* __restrict__ a16, int flip) {
  long idx = (long)blockIdx.x * blockDim.x + threadIdx.x;   // ROWS*H
  if (idx >= (long)CROWS * CH) return;
  int h = idx % CH; long row = idx / CH;
  int t = row % CT; long bm = row / CT;
  int ts = flip ? (CT - 1 - t) : t;
  a16[idx] = (_Float16)xn[((size_t)bm * CT + ts) * CH + h];
}

// ---------------- WMMA GEMM: C[m,n] = sum_k A[m,k] * W[n,k] ----------------
// Template N,K so the K-loop fully unrolls and all guards fold.
// 4 waves / block, 16x48 strip per wave (3 N-tiles share one A fragment).
// 48 divides both N=384 and N=96 -> no per-tile guards, only a wave-uniform
// early-out, so EXEC is all-ones at every WMMA (ISA requirement).
// f16 operand layouts used here (contiguous k, not the ISA table order):
//  A 16x32: lane L holds row (L&15); halves 0-7 -> k = 16*(L>=16)+0..7,
//           halves 8-15 -> k = 16*(L>=16)+8..15      (two 16B loads)
//  B 32x16: lane L holds col (L&15); halves 0-15 -> k = 16*(L>=16)+0..15 (one 32B load)
//  C/D f32: VGPR r -> M = r + 8*(L>=16), N = (L&15)
// Tail M rows: clamp source row (pollutes only never-stored C rows).
template <int N, int K>
__global__ void wmma_gemm_kernel(const _Float16* __restrict__ A, const _Float16* __restrict__ W,
                                 float* __restrict__ C, int M) {
  constexpr int TILES = 3;                        // 48 columns per wave
  static_assert(N % (16 * TILES) == 0, "N must be a multiple of 48");
  int lane = threadIdx.x & 31;
  int wave = threadIdx.x >> 5;
  int m0 = blockIdx.x * 16;
  int nBase = (blockIdx.y * 4 + wave) * (16 * TILES);
  if (nBase >= N) return;                         // wave-uniform
  int laneHi = lane >> 4;
  int lan = lane & 15;
  int rowA = m0 + lan;
  if (rowA > M - 1) rowA = M - 1;                 // clamp tail rows
  const _Float16* Arow = A + (size_t)rowA * K + laneHi * 16;
  const _Float16* Wc[TILES];
#pragma unroll
  for (int tt = 0; tt < TILES; ++tt)
    Wc[tt] = W + (size_t)(nBase + tt * 16 + lan) * K + laneHi * 16;
  v8f acc[TILES] = {};
#pragma unroll
  for (int k0 = 0; k0 < K; k0 += 32) {
    if (k0 + 32 < K) __builtin_prefetch(Arow + k0 + 32, 0, 3);
    v8h alo = *(const v8h*)(Arow + k0);
    v8h ahi = *(const v8h*)(Arow + k0 + 8);
    v16h a = __builtin_shufflevector(alo, ahi, 0, 1, 2, 3, 4, 5, 6, 7,
                                     8, 9, 10, 11, 12, 13, 14, 15);
#pragma unroll
    for (int tt = 0; tt < TILES; ++tt) {
      v16h bmat = *(const v16h*)(Wc[tt] + k0);
      acc[tt] = __builtin_amdgcn_wmma_f32_16x16x32_f16(false, a, false, bmat,
                                                       (short)0, acc[tt], false, false);
    }
  }
#pragma unroll
  for (int tt = 0; tt < TILES; ++tt) {
#pragma unroll
    for (int r8 = 0; r8 < 8; ++r8) {
      int m = m0 + r8 + laneHi * 8;
      if (m < M) C[(size_t)m * N + nBase + tt * 16 + lan] = acc[tt][r8];
    }
  }
}

// ---------------- mamba: causal depthwise conv (KM=4) + SiLU ----------------
__global__ void mconv_kernel(const float* __restrict__ xz, const float* __restrict__ cw,
                             const float* __restrict__ cb, float* __restrict__ xc) {
  long idx = (long)blockIdx.x * blockDim.x + threadIdx.x;   // ROWS*DIN
  if (idx >= (long)CROWS * CDIN) return;
  int c = idx % CDIN; long row = idx / CDIN;
  int t = row % CT; long bm = row / CT;
  float acc = cb[c];
#pragma unroll
  for (int k = 0; k < CKM; ++k) {
    int tt = t + k - (CKM - 1);
    if (tt >= 0) acc += cw[c * CKM + k] * xz[((size_t)bm * CT + tt) * CDP + c];
  }
  xc[idx] = silu_f(acc);
}

__global__ void xproj_kernel(const float* __restrict__ xc, const float* __restrict__ xpw,
                             float* __restrict__ dbl) {
  long idx = (long)blockIdx.x * blockDim.x + threadIdx.x;   // ROWS*XPD
  if (idx >= (long)CROWS * CXPD) return;
  int j = idx % CXPD; long row = idx / CXPD;
  const float* xr = xc + (size_t)row * CDIN;
  const float* wr = xpw + (size_t)j * CDIN;
  float acc = 0.f;
  for (int c = 0; c < CDIN; ++c) acc += xr[c] * wr[c];
  dbl[idx] = acc;
}

__global__ void dt_kernel(const float* __restrict__ dbl, const float* __restrict__ dtw,
                          const float* __restrict__ dtb, float* __restrict__ dt) {
  long idx = (long)blockIdx.x * blockDim.x + threadIdx.x;   // ROWS*DIN
  if (idx >= (long)CROWS * CDIN) return;
  int c = idx % CDIN; long row = idx / CDIN;
  float acc = dtb[c];
#pragma unroll
  for (int r = 0; r < CR; ++r) acc += dbl[row * CXPD + r] * dtw[c * CR + r];
  dt[idx] = acc > 20.f ? acc : log1pf(__expf(acc));   // softplus
}

// ---------------- selective scan: 1 block per sequence, 192 threads ----------------
__global__ void scan_kernel(const float* __restrict__ dt, const float* __restrict__ xc,
                            const float* __restrict__ dbl, const float* __restrict__ A_log,
                            const float* __restrict__ Dp, float* __restrict__ y) {
  __shared__ float sB[CN], sC[CN];
  int bm = blockIdx.x;
  int c = threadIdx.x;                 // 192
  float An[CN], h[CN];
#pragma unroll
  for (int n = 0; n < CN; ++n) { An[n] = -__expf(A_log[c * CN + n]); h[n] = 0.f; }
  float Dc = Dp[c];
  for (int t = 0; t < CT; ++t) {
    size_t row = (size_t)bm * CT + t;
    if (c < CN) sB[c] = dbl[row * CXPD + CR + c];
    else if (c < 2 * CN) sC[c - CN] = dbl[row * CXPD + CR + c];
    __syncthreads();
    float dtv = dt[row * CDIN + c], xv = xc[row * CDIN + c];
    float dx = dtv * xv, acc = 0.f;
#pragma unroll
    for (int n = 0; n < CN; ++n) {
      h[n] = __expf(dtv * An[n]) * h[n] + dx * sB[n];
      acc += h[n] * sC[n];
    }
    y[row * CDIN + c] = acc + xv * Dc;
    __syncthreads();
  }
}

__global__ void gate_kernel(const float* __restrict__ y, const float* __restrict__ xz,
                            _Float16* __restrict__ a16) {
  long idx = (long)blockIdx.x * blockDim.x + threadIdx.x;   // ROWS*DIN
  if (idx >= (long)CROWS * CDIN) return;
  int c = idx % CDIN; long row = idx / CDIN;
  float zv = xz[(size_t)row * CDP + CDIN + c];
  a16[idx] = (_Float16)(y[idx] * silu_f(zv));
}

// ---------------- final combine: x3 + (yf + flip_t(yb)) / 2 ----------------
__global__ void combine_kernel(const float* __restrict__ x3, const float* __restrict__ yf,
                               const float* __restrict__ yb, float* __restrict__ out) {
  long idx = (long)blockIdx.x * blockDim.x + threadIdx.x;   // NTOT
  if (idx >= NTOT) return;
  int h = idx % CH; long r = idx / CH;
  int t = r % CT; long bm = r / CT;     // bm == b*FQ + f
  out[idx] = x3[idx] + 0.5f * (yf[((size_t)bm * CT + t) * CH + h] +
                               yb[((size_t)bm * CT + (CT - 1 - t)) * CH + h]);
}

// ---------------- launch ----------------
extern "C" void kernel_launch(void* const* d_in, const int* in_sizes, int n_in,
                              void* d_out, int out_size, void* d_ws, size_t ws_size,
                              hipStream_t stream) {
  const float* x       = (const float*)d_in[0];
  const float* ln1_g   = (const float*)d_in[1];
  const float* ln1_b   = (const float*)d_in[2];
  const float* conv1_w = (const float*)d_in[3];
  const float* conv1_b = (const float*)d_in[4];
  const float* prelu1  = (const float*)d_in[5];
  const float* lnf_g   = (const float*)d_in[6];
  const float* lnf_b   = (const float*)d_in[7];
  const float* sq_w    = (const float*)d_in[8];
  const float* sq_b    = (const float*)d_in[9];
  const float* full_w  = (const float*)d_in[10];
  const float* full_b  = (const float*)d_in[11];
  const float* unsq_w  = (const float*)d_in[12];
  const float* unsq_b  = (const float*)d_in[13];
  const float* ln2_g   = (const float*)d_in[14];
  const float* ln2_b   = (const float*)d_in[15];
  const float* conv2_w = (const float*)d_in[16];
  const float* conv2_b = (const float*)d_in[17];
  const float* prelu2  = (const float*)d_in[18];
  const float* lnm_g   = (const float*)d_in[19];
  const float* lnm_b   = (const float*)d_in[20];
  float* out = (float*)d_out;

  // workspace carve-out (256B aligned)
  char* wp = (char*)d_ws;
  auto alloc = [&](size_t bytes) -> void* {
    void* p = (void*)wp;
    wp += (bytes + 255) & ~(size_t)255;
    return p;
  };
  float* yln   = (float*)alloc(NTOT * 4);
  float* x1    = (float*)alloc(NTOT * 4);
  float* x2    = (float*)alloc(NTOT * 4);
  float* x3    = (float*)alloc(NTOT * 4);
  float* xn    = (float*)alloc(NTOT * 4);
  float* zbuf  = (float*)alloc((size_t)CB * CT * CS * CFQ * 4);
  float* wbuf  = (float*)alloc((size_t)CB * CT * CS * CFQ * 4);
  _Float16* a16in = (_Float16*)alloc((size_t)CROWS * CH * 2);
  _Float16* a16g  = (_Float16*)alloc((size_t)CROWS * CDIN * 2);
  _Float16* w16in = (_Float16*)alloc((size_t)CDP * CH * 2);
  _Float16* w16out= (_Float16*)alloc((size_t)CH * CDIN * 2);
  float* xz    = (float*)alloc((size_t)CROWS * CDP * 4);
  float* xc    = (float*)alloc((size_t)CROWS * CDIN * 4);
  float* dblb  = (float*)alloc((size_t)CROWS * CXPD * 4);
  float* dtb   = (float*)alloc((size_t)CROWS * CDIN * 4);
  float* ybuf  = (float*)alloc((size_t)CROWS * CDIN * 4);
  float* yout0 = (float*)alloc((size_t)CROWS * CH * 4);
  float* yout1 = (float*)alloc((size_t)CROWS * CH * 4);

  auto cdiv = [](long a, long b) { return (int)((a + b - 1) / b); };
  const int rowsBFT = CB * CFQ * CT;   // 51400

  // block 1: x1 = x + fconv(x)
  ln_kernel<<<rowsBFT, 32, 0, stream>>>(x, ln1_g, ln1_b, yln, rowsBFT);
  fconv_kernel<<<rowsBFT, CH, 0, stream>>>(x, yln, conv1_w, conv1_b, prelu1, x1);

  // block 2: x2 = x1 + full(x1)
  ln_kernel<<<rowsBFT, 32, 0, stream>>>(x1, lnf_g, lnf_b, yln, rowsBFT);
  {
    long n = (long)CB * CT * CS * CFQ;
    squeeze_kernel<<<cdiv(n, 256), 256, 0, stream>>>(yln, sq_w, sq_b, zbuf);
    fullmm_kernel<<<cdiv(n, 256), 256, 0, stream>>>(zbuf, full_w, full_b, wbuf);
  }
  unsqueeze_kernel<<<cdiv(NTOT, 256), 256, 0, stream>>>(x1, wbuf, unsq_w, unsq_b, x2);

  // block 3: x3 = x2 + fconv(x2)
  ln_kernel<<<rowsBFT, 32, 0, stream>>>(x2, ln2_g, ln2_b, yln, rowsBFT);
  fconv_kernel<<<rowsBFT, CH, 0, stream>>>(x2, yln, conv2_w, conv2_b, prelu2, x3);

  // mamba input LN
  ln_kernel<<<rowsBFT, 32, 0, stream>>>(x3, lnm_g, lnm_b, xn, rowsBFT);

  for (int d = 0; d < 2; ++d) {
    const float* in_w   = (const float*)d_in[21 + 9 * d];
    const float* mcw    = (const float*)d_in[22 + 9 * d];
    const float* mcb    = (const float*)d_in[23 + 9 * d];
    const float* xp_w   = (const float*)d_in[24 + 9 * d];
    const float* dt_w   = (const float*)d_in[25 + 9 * d];
    const float* dt_bi  = (const float*)d_in[26 + 9 * d];
    const float* A_log  = (const float*)d_in[27 + 9 * d];
    const float* Dp     = (const float*)d_in[28 + 9 * d];
    const float* out_w  = (const float*)d_in[29 + 9 * d];
    float* youtd = d == 0 ? yout0 : yout1;

    cvt16_kernel<<<cdiv((long)CDP * CH, 256), 256, 0, stream>>>(in_w, w16in, (long)CDP * CH);
    cvt16_kernel<<<cdiv((long)CH * CDIN, 256), 256, 0, stream>>>(out_w, w16out, (long)CH * CDIN);
    prep_inA_kernel<<<cdiv((long)CROWS * CH, 256), 256, 0, stream>>>(xn, a16in, d);

    // in_proj: [51400 x 96] * [96 x 384]^T  -> WMMA (16x48 per wave)
    {
      dim3 grid(cdiv(CROWS, 16), cdiv(CDP, 192));
      wmma_gemm_kernel<CDP, CH><<<grid, 128, 0, stream>>>(a16in, w16in, xz, CROWS);
    }
    mconv_kernel<<<cdiv((long)CROWS * CDIN, 256), 256, 0, stream>>>(xz, mcw, mcb, xc);
    xproj_kernel<<<cdiv((long)CROWS * CXPD, 256), 256, 0, stream>>>(xc, xp_w, dblb);
    dt_kernel<<<cdiv((long)CROWS * CDIN, 256), 256, 0, stream>>>(dblb, dt_w, dt_bi, dtb);
    scan_kernel<<<CBM, CDIN, 0, stream>>>(dtb, xc, dblb, A_log, Dp, ybuf);
    gate_kernel<<<cdiv((long)CROWS * CDIN, 256), 256, 0, stream>>>(ybuf, xz, a16g);

    // out_proj: [51400 x 192] * [192 x 96]^T -> WMMA
    {
      dim3 grid(cdiv(CROWS, 16), cdiv(CH, 192));
      wmma_gemm_kernel<CH, CDIN><<<grid, 128, 0, stream>>>(a16g, w16out, youtd, CROWS);
    }
  }

  combine_kernel<<<cdiv(NTOT, 256), 256, 0, stream>>>(x3, yout0, yout1, out);
}
